// DeformableConvBlock_79233556676779
// MI455X (gfx1250) — matrix operand order repaired and checked by reading, not compiled, not run on hardware
//
#include <hip/hip_runtime.h>
#include <hip/hip_bf16.h>

typedef __attribute__((ext_vector_type(16))) __bf16 v16bf;
typedef __attribute__((ext_vector_type(8)))  __bf16 v8bf;
typedef __attribute__((ext_vector_type(8)))  float  v8f;
typedef __attribute__((ext_vector_type(4)))  float  v4f;

#define Bsz 4
#define Cc  128
#define Hh  96
#define Ww  96
#define HW  (Hh * Ww)      // 9216
#define C2  (2 * Cc)       // 256
#define K1  (C2 * 9)       // 2304  (conv1 GEMM K, tap-major: kk = tap*256 + c)
#define KD  (Cc * 9)       // 1152  (deform GEMM K, tap-major: kk = tap*128 + c)

union PK4 { __bf16 h[4]; unsigned long long u; };

// ---------------------------------------------------------------------------
// Weight prep (tap-major re-order, fp32 -> bf16):
//   w1[o][tap*256 + c] = est_w[o][c][tap]
//   w2[o][tap*128 + c] = dc_w [o][c][tap]
// ---------------------------------------------------------------------------
__global__ __launch_bounds__(256)
void prep_weights(const float* __restrict__ est_w, const float* __restrict__ dc_w,
                  __bf16* __restrict__ w1_bf, __bf16* __restrict__ w2_bf) {
    int idx = blockIdx.x * 256 + threadIdx.x;
    const int n1 = C2 * K1;   // 589824
    const int n2 = Cc * KD;   // 147456
    if (idx < n1) {
        int o   = idx / K1;
        int kk  = idx - o * K1;
        int tap = kk >> 8;
        int c   = kk & 255;
        w1_bf[idx] = (__bf16)est_w[(o * C2 + c) * 9 + tap];
    } else if (idx < n1 + n2) {
        int j   = idx - n1;
        int o   = j / KD;
        int kk  = j - o * KD;
        int tap = kk >> 7;
        int c   = kk & 127;
        w2_bf[j] = (__bf16)dc_w[(o * Cc + c) * 9 + tap];
    }
}

// ---------------------------------------------------------------------------
// conv1: est = relu(conv3x3(concat(lr,hr)) + est_b)
// Implicit GEMM M=36864, N=256, K=2304 (tap-major). WG = 8 waves.
// WG tile 32M x 256N; wave = 16M x 64N -> 4 WMMA per 32-K step.
// Double-buffered LDS A; packed b64 staging (4 channels / thread).
// ---------------------------------------------------------------------------
__global__ __launch_bounds__(256)
void conv1_wmma(const float* __restrict__ lr, const float* __restrict__ hr,
                const __bf16* __restrict__ wbf, const float* __restrict__ bias,
                float* __restrict__ est) {
    __shared__ __bf16 sA[2][32][32];         // 4 KB double-buffered

    const int tid   = threadIdx.x;
    const int lane  = tid & 31;
    const int wv    = tid >> 5;
    const int mw    = wv & 1;                // M subtile
    const int nw    = wv >> 1;               // N quadrant (64 wide)
    const int P     = blockIdx.x * 32;       // 32 consecutive x in one row
    const int b     = P / HW;
    const int rem   = P - b * HW;
    const int y     = rem / Ww;
    const int xbase = rem - y * Ww;
    const int lm    = lane & 15;
    const int lh    = lane >> 4;

    const float* lrb = lr + (size_t)b * Cc * HW;
    const float* hrb = hr + (size_t)b * Cc * HW;

    // staging: thread owns (pixel row, 4 consecutive channels) -> 1 ds_store_b64
    const int srow = tid >> 3;               // pixel 0..31
    const int scl  = (tid & 7) * 4;          // channel group 0,4,...,28

    auto stage = [&](int ks, int buf) {
        int tap   = ks >> 8;
        int cbase = ks & 255;
        int t3    = tap / 3;
        int dy    = t3 - 1;
        int dx    = tap - 3 * t3 - 1;
        int ys    = y + dy;
        int ysc   = ys < 0 ? 0 : (ys > Hh - 1 ? Hh - 1 : ys);
        int xs    = xbase + srow + dx;
        int xsc   = xs < 0 ? 0 : (xs > Ww - 1 ? Ww - 1 : xs);
        bool ok   = ((unsigned)ys < (unsigned)Hh) && ((unsigned)xs < (unsigned)Ww);
        int  sp   = ysc * Ww + xsc;
        int  c    = cbase + scl;             // 4-group never crosses the lr/hr split
        const float* src = (c < Cc) ? (lrb + (size_t)c * HW)
                                    : (hrb + (size_t)(c - Cc) * HW);
        PK4 pk;
        #pragma unroll
        for (int q = 0; q < 4; ++q)
            pk.h[q] = (__bf16)src[(size_t)q * HW + sp];
        unsigned long long r = ok ? pk.u : 0ull;
        *(unsigned long long*)&sA[buf][srow][scl] = r;
    };

    v8f acc0 = {}, acc1 = {}, acc2 = {}, acc3 = {};
    const __bf16* bp0 = wbf + (size_t)(nw * 64 +  0 + lm) * K1 + 16 * lh;
    const __bf16* bp1 = wbf + (size_t)(nw * 64 + 16 + lm) * K1 + 16 * lh;
    const __bf16* bp2 = wbf + (size_t)(nw * 64 + 32 + lm) * K1 + 16 * lh;
    const __bf16* bp3 = wbf + (size_t)(nw * 64 + 48 + lm) * K1 + 16 * lh;

    stage(0, 0);
    const int NSTEP = K1 / 32;               // 72
    for (int s = 0; s < NSTEP; ++s) {
        __syncthreads();
        if (s + 1 < NSTEP) stage((s + 1) * 32, (s + 1) & 1);

        const int ks  = s * 32;
        const int buf = s & 1;
        const int arow = 16 * mw + lm;
        v8bf alo = *(const v8bf*)(&sA[buf][arow][16 * lh]);
        v8bf ahi = *(const v8bf*)(&sA[buf][arow][16 * lh + 8]);
        v16bf a  = __builtin_shufflevector(alo, ahi,
                       0,1,2,3,4,5,6,7,8,9,10,11,12,13,14,15);

        __builtin_prefetch((const void*)(bp0 + ks + 32), 0, 1);

        #define B_FRAG(bp)  __builtin_shufflevector(                       \
                *(const v8bf*)((bp) + ks), *(const v8bf*)((bp) + ks + 8),  \
                0,1,2,3,4,5,6,7,8,9,10,11,12,13,14,15)
        v16bf b0 = B_FRAG(bp0);
        v16bf b1 = B_FRAG(bp1);
        v16bf b2 = B_FRAG(bp2);
        v16bf b3 = B_FRAG(bp3);
        #undef B_FRAG

        acc0 = __builtin_amdgcn_wmma_f32_16x16x32_bf16(false, a, false, b0, (short)0, acc0, false, false);
        acc1 = __builtin_amdgcn_wmma_f32_16x16x32_bf16(false, a, false, b1, (short)0, acc1, false, false);
        acc2 = __builtin_amdgcn_wmma_f32_16x16x32_bf16(false, a, false, b2, (short)0, acc2, false, false);
        acc3 = __builtin_amdgcn_wmma_f32_16x16x32_bf16(false, a, false, b3, (short)0, acc3, false, false);
    }

    // epilogue: bias + ReLU, vectorized stores (8 consecutive x per subtile)
    const int m0 = 16 * mw + 8 * lh;
    #pragma unroll
    for (int nn = 0; nn < 4; ++nn) {
        v8f acc = nn == 0 ? acc0 : nn == 1 ? acc1 : nn == 2 ? acc2 : acc3;
        const int   o  = nw * 64 + nn * 16 + lm;
        const float bo = bias[o];
        float* dst = est + (size_t)(b * C2 + o) * HW + y * Ww + xbase + m0;
        v4f f0, f1;
        #pragma unroll
        for (int r = 0; r < 4; ++r) {
            float u = acc[r] + bo;     f0[r] = u > 0.0f ? u : 0.0f;
            float v = acc[r + 4] + bo; f1[r] = v > 0.0f ? v : 0.0f;
        }
        *(v4f*)(dst)     = f0;
        *(v4f*)(dst + 4) = f1;
    }
}

// ---------------------------------------------------------------------------
// conv2: offsets = conv3x3(est) + off_b  (N=18, 3 GFLOP -> VALU, branchless)
// ---------------------------------------------------------------------------
__global__ __launch_bounds__(256)
void conv2_offsets(const float* __restrict__ est, const float* __restrict__ ow,
                   const float* __restrict__ ob, float* __restrict__ offs) {
    const int bo = blockIdx.x;               // b*18 + o
    const int b  = bo / 18;
    const int o  = bo - b * 18;
    const int p  = blockIdx.y * 256 + threadIdx.x;
    const int y  = p / Ww;
    const int x  = p - y * Ww;

    float acc = ob[o];
    const float* wrow = ow + (size_t)o * C2 * 9;
    const float* eb   = est + (size_t)b * C2 * HW;
    for (int c = 0; c < C2; ++c) {
        const float* ec = eb + (size_t)c * HW;
        const float* wc = wrow + c * 9;
        #pragma unroll
        for (int t = 0; t < 9; ++t) {
            int dy = t / 3 - 1, dx = t % 3 - 1;
            int ys = y + dy, xs = x + dx;
            int ysc = ys < 0 ? 0 : (ys > Hh - 1 ? Hh - 1 : ys);
            int xsc = xs < 0 ? 0 : (xs > Ww - 1 ? Ww - 1 : xs);
            bool ok = ((unsigned)ys < (unsigned)Hh) && ((unsigned)xs < (unsigned)Ww);
            float v = ec[ysc * Ww + xsc];
            acc += (ok ? v : 0.0f) * wc[t];
        }
    }
    offs[(size_t)bo * HW + p] = acc;
}

// ---------------------------------------------------------------------------
// Deformable conv: implicit GEMM. WG: 32 pixels x 128N, K = 1152 (tap-major).
// 8 waves; wave = 16M x 32N -> 2 WMMA per 32-K step (one tap, 32 channels).
// Bilinear meta cached in LDS; packed b64 staging; double-buffered A tile.
// ---------------------------------------------------------------------------
__global__ __launch_bounds__(256)
void deform_wmma(const float* __restrict__ hr, const float* __restrict__ offs,
                 const __bf16* __restrict__ w2, const float* __restrict__ bias,
                 float* __restrict__ out) {
    __shared__ __bf16 sA[2][32][32];         // 4 KB double-buffered
    __shared__ int    sO00[288], sO01[288], sO10[288], sO11[288];
    __shared__ float  sW00[288], sW01[288], sW10[288], sW11[288];

    const int tid   = threadIdx.x;
    const int lane  = tid & 31;
    const int wv    = tid >> 5;
    const int mw    = wv & 1;
    const int nw    = wv >> 1;
    const int P     = blockIdx.x * 32;
    const int b     = P / HW;
    const int rem   = P - b * HW;
    const int y     = rem / Ww;
    const int xbase = rem - y * Ww;
    const int lm    = lane & 15;
    const int lh    = lane >> 4;

    const float* hrb = hr + (size_t)b * Cc * HW;

    // ---- phase 0: bilinear corner *offsets* + validity-folded weights ----
    for (int mi = tid; mi < 288; mi += 256) {
        int p = mi / 9;
        int k = mi - p * 9;
        int x = xbase + p;
        float dy = offs[(size_t)(b * 18 + 2 * k)     * HW + y * Ww + x];
        float dx = offs[(size_t)(b * 18 + 2 * k + 1) * HW + y * Ww + x];
        float py = (float)(y + (k / 3) - 1) + dy;
        float px = (float)(x + (k % 3) - 1) + dx;
        float fy = floorf(py), fx = floorf(px);
        float ly = py - fy,    lx = px - fx;
        int y0 = (int)fy, x0 = (int)fx;
        int y1 = y0 + 1,  x1 = x0 + 1;
        float vy0 = ((unsigned)y0 < (unsigned)Hh) ? 1.f : 0.f;
        float vy1 = ((unsigned)y1 < (unsigned)Hh) ? 1.f : 0.f;
        float vx0 = ((unsigned)x0 < (unsigned)Ww) ? 1.f : 0.f;
        float vx1 = ((unsigned)x1 < (unsigned)Ww) ? 1.f : 0.f;
        sW00[mi] = (1.f - ly) * (1.f - lx) * vy0 * vx0;
        sW01[mi] = (1.f - ly) * lx         * vy0 * vx1;
        sW10[mi] = ly         * (1.f - lx) * vy1 * vx0;
        sW11[mi] = ly * lx                 * vy1 * vx1;
        int y0c = y0 < 0 ? 0 : (y0 > Hh - 1 ? Hh - 1 : y0);
        int y1c = y1 < 0 ? 0 : (y1 > Hh - 1 ? Hh - 1 : y1);
        int x0c = x0 < 0 ? 0 : (x0 > Ww - 1 ? Ww - 1 : x0);
        int x1c = x1 < 0 ? 0 : (x1 > Ww - 1 ? Ww - 1 : x1);
        sO00[mi] = y0c * Ww + x0c;
        sO01[mi] = y0c * Ww + x1c;
        sO10[mi] = y1c * Ww + x0c;
        sO11[mi] = y1c * Ww + x1c;
    }
    __syncthreads();

    // staging: thread owns (pixel, 4 consecutive channels) -> 1 ds_store_b64
    const int srow = tid >> 3;               // pixel 0..31
    const int scl  = (tid & 7) * 4;          // channel group

    auto stage = [&](int s, int buf) {
        int tap   = s >> 2;
        int cbase = (s & 3) * 32;
        int mi    = srow * 9 + tap;
        int   o00 = sO00[mi], o01 = sO01[mi], o10 = sO10[mi], o11 = sO11[mi];
        float w00 = sW00[mi], w01 = sW01[mi], w10 = sW10[mi], w11 = sW11[mi];
        const float* src = hrb + (size_t)(cbase + scl) * HW;
        PK4 pk;
        #pragma unroll
        for (int q = 0; q < 4; ++q) {
            const float* sc = src + (size_t)q * HW;
            float v = w00 * sc[o00] + w01 * sc[o01]
                    + w10 * sc[o10] + w11 * sc[o11];
            pk.h[q] = (__bf16)v;
        }
        *(unsigned long long*)&sA[buf][srow][scl] = pk.u;
    };

    v8f acc0 = {}, acc1 = {};
    const __bf16* bp0 = w2 + (size_t)(nw * 32 +  0 + lm) * KD + 16 * lh;
    const __bf16* bp1 = w2 + (size_t)(nw * 32 + 16 + lm) * KD + 16 * lh;

    stage(0, 0);
    const int NSTEP = KD / 32;               // 36
    for (int s = 0; s < NSTEP; ++s) {
        __syncthreads();
        if (s + 1 < NSTEP) stage(s + 1, (s + 1) & 1);

        const int kk  = s * 32;
        const int buf = s & 1;
        const int arow = 16 * mw + lm;
        v8bf alo = *(const v8bf*)(&sA[buf][arow][16 * lh]);
        v8bf ahi = *(const v8bf*)(&sA[buf][arow][16 * lh + 8]);
        v16bf a  = __builtin_shufflevector(alo, ahi,
                       0,1,2,3,4,5,6,7,8,9,10,11,12,13,14,15);

        __builtin_prefetch((const void*)(bp0 + kk + 32), 0, 1);

        v16bf b0 = __builtin_shufflevector(
            *(const v8bf*)(bp0 + kk), *(const v8bf*)(bp0 + kk + 8),
            0,1,2,3,4,5,6,7,8,9,10,11,12,13,14,15);
        v16bf b1 = __builtin_shufflevector(
            *(const v8bf*)(bp1 + kk), *(const v8bf*)(bp1 + kk + 8),
            0,1,2,3,4,5,6,7,8,9,10,11,12,13,14,15);

        acc0 = __builtin_amdgcn_wmma_f32_16x16x32_bf16(false, a, false, b0, (short)0, acc0, false, false);
        acc1 = __builtin_amdgcn_wmma_f32_16x16x32_bf16(false, a, false, b1, (short)0, acc1, false, false);
    }

    const int m0 = 16 * mw + 8 * lh;
    #pragma unroll
    for (int nn = 0; nn < 2; ++nn) {
        v8f acc = nn == 0 ? acc0 : acc1;
        const int   o  = nw * 32 + nn * 16 + lm;
        const float bo = bias[o];
        float* dst = out + (size_t)(b * Cc + o) * HW + y * Ww + xbase + m0;
        v4f f0, f1;
        #pragma unroll
        for (int r = 0; r < 4; ++r) {
            f0[r] = acc[r] + bo;
            f1[r] = acc[r + 4] + bo;
        }
        *(v4f*)(dst)     = f0;
        *(v4f*)(dst + 4) = f1;
    }
}

// ---------------------------------------------------------------------------
// Launch
// ---------------------------------------------------------------------------
extern "C" void kernel_launch(void* const* d_in, const int* in_sizes, int n_in,
                              void* d_out, int out_size, void* d_ws, size_t ws_size,
                              hipStream_t stream) {
    (void)in_sizes; (void)n_in; (void)out_size; (void)ws_size;

    const float* lr   = (const float*)d_in[0];
    const float* hr   = (const float*)d_in[1];
    const float* estw = (const float*)d_in[2];
    const float* estb = (const float*)d_in[3];
    const float* offw = (const float*)d_in[4];
    const float* offb = (const float*)d_in[5];
    const float* dcw  = (const float*)d_in[6];
    const float* dcb  = (const float*)d_in[7];
    float* out = (float*)d_out;

    char* ws = (char*)d_ws;
    float*  est     = (float*)ws;                       // 37,748,736 B
    float*  offsbuf = (float*)(ws + 37748736);          //  2,654,208 B
    __bf16* w1_bf   = (__bf16*)(ws + 40402944);         //  1,179,648 B
    __bf16* w2_bf   = (__bf16*)(ws + 41582592);         //    294,912 B

    prep_weights<<<2880, 256, 0, stream>>>(estw, dcw, w1_bf, w2_bf);
    conv1_wmma  <<<1152, 256, 0, stream>>>(lr, hr, w1_bf, estb, est);
    conv2_offsets<<<dim3(72, 36), 256, 0, stream>>>(est, offw, offb, offsbuf);
    deform_wmma <<<1152, 256, 0, stream>>>(hr, offsbuf, w2_bf, dcb, out);
}
